// HGT_17901423690440
// MI455X (gfx1250) — matrix operand order repaired and checked by reading, not compiled, not run on hardware
//
#include <hip/hip_runtime.h>
#include <hip/hip_bf16.h>

// ---------------------------------------------------------------------------
// HGT forward for MI455X (gfx1250, wave32, WMMA).
// All GEMM-like math uses v_wmma_f32_16x16x32_bf16 (fp32 accumulate).
// bf16 conversion is native (fptrunc -> hardware cvt), not software RNE.
// ---------------------------------------------------------------------------

typedef __attribute__((ext_vector_type(16))) __bf16 v16bf;
typedef __attribute__((ext_vector_type(8)))  float  v8f;

#define DEV static __device__ __forceinline__

constexpr int N_ = 20000;
constexpr int E_ = 320000;
constexpr int D_ = 256;
constexpr int H_ = 8;
constexpr int T_ = 3;
constexpr int R_ = 5;
constexpr int C_ = 16;

// ---- workspace layout (bytes) ----------------------------------------------
constexpr size_t SZ_WT   = (size_t)4 * T_ * D_ * D_ * 2;        // bf16 [mat][t][n][k]
constexpr size_t SZ_REL  = (size_t)R_ * H_ * 32 * 32 * 2;       // bf16 [r][h][l][k]
constexpr size_t SZ_WOUT = (size_t)C_ * D_ * 2;                 // bf16 [c][k]
constexpr size_t OFF_WT   = 0;
constexpr size_t OFF_RELA = OFF_WT + SZ_WT;
constexpr size_t OFF_RELM = OFF_RELA + SZ_REL;
constexpr size_t OFF_WOUT = OFF_RELM + SZ_REL;
constexpr size_t OFF_KBF  = OFF_WOUT + SZ_WOUT;                 // bf16 [N][256]
constexpr size_t OFF_VBF  = OFF_KBF + (size_t)N_ * D_ * 2;      // bf16 [N][256]
constexpr size_t OFF_QF   = OFF_VBF + (size_t)N_ * D_ * 2;      // f32 [N][256]
constexpr size_t OFF_EXL  = OFF_QF  + (size_t)N_ * D_ * 4;      // f32 [E][8]
constexpr size_t OFF_MX   = OFF_EXL + (size_t)E_ * H_ * 4;      // f32 [N][8]
constexpr size_t OFF_SM   = OFF_MX  + (size_t)N_ * H_ * 4;      // f32 [N][8]
constexpr size_t OFF_AGGR = OFF_SM  + (size_t)N_ * H_ * 4;      // f32 [N][256]
constexpr size_t OFF_HBF  = OFF_AGGR + (size_t)N_ * D_ * 4;     // bf16 [N][256]

// ---- helpers ---------------------------------------------------------------
DEV v16bf zero_frag() {
  v16bf z;
#pragma unroll
  for (int i = 0; i < 16; ++i) z[i] = (__bf16)0.0f;
  return z;
}
// A fragment, 16x32 bf16 tile (ISA 7.12.2): lane m=L&15, half=L>>4.
// element i<8  -> K = half*8 + i ; element i>=8 -> K = half*8 + i + 8
DEV v16bf load_a_bf(const __bf16* chunk, int half) {
  v16bf a; const __bf16* p = chunk + half * 8;
#pragma unroll
  for (int i = 0; i < 8; ++i)  a[i] = p[i];
#pragma unroll
  for (int i = 8; i < 16; ++i) a[i] = p[i + 8];
  return a;
}
DEV v16bf load_a_f32(const float* chunk, int half) {
  v16bf a; const float* p = chunk + half * 8;
#pragma unroll
  for (int i = 0; i < 8; ++i)  a[i] = (__bf16)p[i];
#pragma unroll
  for (int i = 8; i < 16; ++i) a[i] = (__bf16)p[i + 8];
  return a;
}
DEV v16bf load_a_elu(const float* chunk, int half) {
  v16bf a; const float* p = chunk + half * 8;
#pragma unroll
  for (int i = 0; i < 8; ++i) {
    float v = p[i]; v = v > 0.f ? v : __expf(v) - 1.f;
    a[i] = (__bf16)v;
  }
#pragma unroll
  for (int i = 8; i < 16; ++i) {
    float v = p[i + 8]; v = v > 0.f ? v : __expf(v) - 1.f;
    a[i] = (__bf16)v;
  }
  return a;
}
// B fragment, 32x16 bf16: lane col = L&15, K_i = half*8 + i (i<8), half*8 + i + 8 (i>=8);
// K values for one column are contiguous in memory (weights pre-transposed to [n][k]).
DEV v16bf load_b_bf(const __bf16* colptr, int half) {
  v16bf b; const __bf16* p = colptr + half * 8;
#pragma unroll
  for (int i = 0; i < 8; ++i)  b[i] = p[i];
#pragma unroll
  for (int i = 8; i < 16; ++i) b[i] = p[i + 8];
  return b;
}
DEV v8f wmma_bf16(v16bf a, v16bf b, v8f c) {
  return __builtin_amdgcn_wmma_f32_16x16x32_bf16(false, a, false, b, (short)0, c,
                                                 false, false);
}

// ---- kernel 0: weight repack (fp32 -> bf16, transposed for B fragments) ----
__global__ void pack_kernel(const float* __restrict__ Wk, const float* __restrict__ Wq,
                            const float* __restrict__ Wv, const float* __restrict__ Wa,
                            const float* __restrict__ rel_att,
                            const float* __restrict__ rel_msg,
                            const float* __restrict__ W_out,
                            __bf16* __restrict__ WT,
                            __bf16* __restrict__ relAT,
                            __bf16* __restrict__ relMT,
                            __bf16* __restrict__ WoutT) {
  const int NW   = 4 * T_ * D_ * D_;            // 786432
  const int NREL = R_ * H_ * 32 * 32;           // 40960
  const int total = NW + 2 * NREL + C_ * D_;
  const float* Ws[4] = {Wk, Wq, Wv, Wa};
  for (int idx = blockIdx.x * blockDim.x + threadIdx.x; idx < total;
       idx += gridDim.x * blockDim.x) {
    if (idx < NW) {
      int i = idx;
      int k = i & 255; i >>= 8;
      int n = i & 255; i >>= 8;
      int t = i % 3, mat = i / 3;
      WT[idx] = (__bf16)Ws[mat][((size_t)t * D_ + k) * D_ + n];
    } else if (idx < NW + NREL) {
      int i = idx - NW;                          // [r][h][l][k] <- src [rh][k][l]
      int k = i & 31; int l = (i >> 5) & 31; int rh = i >> 10;
      relAT[i] = (__bf16)rel_att[((size_t)rh * 32 + k) * 32 + l];
    } else if (idx < NW + 2 * NREL) {
      int i = idx - NW - NREL;
      int k = i & 31; int l = (i >> 5) & 31; int rh = i >> 10;
      relMT[i] = (__bf16)rel_msg[((size_t)rh * 32 + k) * 32 + l];
    } else {
      int i = idx - NW - 2 * NREL;               // [n][k] <- src [k][n]
      int k = i & 255; int n = i >> 8;
      WoutT[i] = (__bf16)W_out[(size_t)k * C_ + n];
    }
  }
}

// ---- kernel 1: init softmax / aggregation buffers --------------------------
__global__ void init_kernel(float* __restrict__ mx, float* __restrict__ sm,
                            float* __restrict__ aggr) {
  int idx = blockIdx.x * blockDim.x + threadIdx.x;
  if (idx < N_ * D_) aggr[idx] = 0.f;
  if (idx < N_ * H_) { mx[idx] = -3.0e38f; sm[idx] = 0.f; }
}

// ---- kernel 2: typed K/Q/V projections (type-masked WMMA accumulation) -----
__global__ void __launch_bounds__(256)
qkv_kernel(const float* __restrict__ x, const int* __restrict__ ntype,
           const __bf16* __restrict__ WT,
           const float* __restrict__ bk, const float* __restrict__ bq,
           const float* __restrict__ bv,
           __bf16* __restrict__ Kbf, __bf16* __restrict__ Vbf,
           float* __restrict__ Qf) {
  __shared__ int sty[16];
  const int tile = blockIdx.x, tid = threadIdx.x;
  if (tid < 16) sty[tid] = ntype[tile * 16 + tid];
  __syncthreads();
  const int wave = tid >> 5, lane = tid & 31;
  const int m = lane & 15, half = lane >> 4;
  const int row = tile * 16 + m;
  const int tym = sty[m];
  v16bf afrag[8];
#pragma unroll
  for (int kc = 0; kc < 8; ++kc)
    afrag[kc] = load_a_f32(x + (size_t)row * D_ + kc * 32, half);
  const v16bf za = zero_frag();

#pragma unroll 1
  for (int j = 0; j < 6; ++j) {                  // 48 (proj,ntile) pairs / 8 waves
    const int pid = wave * 6 + j;
    const int p = pid >> 4, nt = pid & 15;       // p: 0=K 1=Q 2=V
    const int col = nt * 16 + m;
    const __bf16* Wb = WT + (size_t)p * 3 * 65536;
    v8f c = {};
    for (int t = 0; t < T_; ++t) {
#pragma unroll
      for (int kc = 0; kc < 8; ++kc) {
        v16bf a2 = (tym == t) ? afrag[kc] : za;
        v16bf b = load_b_bf(Wb + (size_t)t * 65536 + (size_t)col * 256 + kc * 32, half);
        c = wmma_bf16(a2, b, c);
      }
    }
    const float* bias = (p == 0) ? bk : (p == 1) ? bq : bv;
#pragma unroll
    for (int jj = 0; jj < 8; ++jj) {             // C layout: n=lane&15, m=half*8+jj
      int mr = half * 8 + jj;
      int grow = tile * 16 + mr;
      float val = c[jj] + bias[(size_t)sty[mr] * D_ + col];
      if (p == 0)      Kbf[(size_t)grow * D_ + col] = (__bf16)val;
      else if (p == 1) Qf[(size_t)grow * D_ + col] = val;
      else             Vbf[(size_t)grow * D_ + col] = (__bf16)val;
    }
  }
}

// ---- kernel 3: per-edge relation bilinear (Kt) + attention logits ----------
__global__ void __launch_bounds__(256)
edge_logits_kernel(const int* __restrict__ src, const int* __restrict__ dst,
                   const int* __restrict__ etype,
                   const __bf16* __restrict__ Kbf,
                   const float* __restrict__ Qf,
                   const __bf16* __restrict__ relAT,
                   const float* __restrict__ rel_prior,
                   float* __restrict__ logits) {
  __shared__ int ssrc[16], sdst[16], setp[16];
  __shared__ float skt[H_][16][32];
  const int tile = blockIdx.x, tid = threadIdx.x;
  if (tid < 16) {
    int e = tile * 16 + tid;
    ssrc[tid] = src[e]; sdst[tid] = dst[e]; setp[tid] = etype[e];
  }
  __syncthreads();
  const int h = tid >> 5, lane = tid & 31;
  const int m = lane & 15, half = lane >> 4;
  const int et = setp[m];
  v16bf a = load_a_bf(Kbf + (size_t)ssrc[m] * D_ + h * 32, half);
  const v16bf za = zero_frag();
  v8f c0 = {}, c1 = {};
  for (int r = 0; r < R_; ++r) {                 // relation-masked accumulation
    v16bf am = (et == r) ? a : za;
    const __bf16* b0p = relAT + (((size_t)(r * H_ + h) * 32) + m) * 32;
    v16bf b0 = load_b_bf(b0p, half);             // l = 0..15
    v16bf b1 = load_b_bf(b0p + 16 * 32, half);   // l = 16..31
    c0 = wmma_bf16(am, b0, c0);
    c1 = wmma_bf16(am, b1, c1);
  }
#pragma unroll
  for (int jj = 0; jj < 8; ++jj) {
    int mr = half * 8 + jj;
    skt[h][mr][m] = c0[jj];
    skt[h][mr][16 + m] = c1[jj];
  }
  __syncthreads();
  if (lane < 16) {                               // logits = (Qe . Kt) * prior / sqrt(K)
    int e = tile * 16 + lane;
    const float* q = Qf + (size_t)sdst[lane] * D_ + h * 32;
    float acc = 0.f;
#pragma unroll
    for (int l = 0; l < 32; ++l) acc += q[l] * skt[h][lane][l];
    logits[(size_t)e * H_ + h] =
        acc * rel_prior[setp[lane] * H_ + h] * 0.17677669529663687f;
  }
}

// ---- kernel 4: segment max (float atomic max via int/uint trick) -----------
__global__ void segmax_kernel(const float* __restrict__ logits,
                              const int* __restrict__ dst, float* __restrict__ mx) {
  int idx = blockIdx.x * blockDim.x + threadIdx.x;
  if (idx >= E_ * H_) return;
  int e = idx >> 3, h = idx & 7;
  float v = logits[idx];
  float* addr = &mx[(size_t)dst[e] * H_ + h];
  if (v >= 0.f) atomicMax((int*)addr, __float_as_int(v));
  else          atomicMin((unsigned int*)addr, __float_as_uint(v));
}

// ---- kernel 5: exp + segment sum (in-place over logits buffer) -------------
__global__ void segexp_kernel(float* __restrict__ exl, const int* __restrict__ dst,
                              const float* __restrict__ mx, float* __restrict__ sm) {
  int idx = blockIdx.x * blockDim.x + threadIdx.x;
  if (idx >= E_ * H_) return;
  int e = idx >> 3, h = idx & 7;
  int d = dst[e];
  float ex = __expf(exl[idx] - mx[(size_t)d * H_ + h]);
  exl[idx] = ex;
  atomicAdd(&sm[(size_t)d * H_ + h], ex);
}

// ---- kernel 6: per-edge message bilinear (Mt) + weighted scatter-add -------
__global__ void __launch_bounds__(256)
edge_msg_kernel(const int* __restrict__ src, const int* __restrict__ dst,
                const int* __restrict__ etype,
                const __bf16* __restrict__ Vbf,
                const __bf16* __restrict__ relMT,
                const float* __restrict__ exl, const float* __restrict__ sm,
                float* __restrict__ aggr) {
  __shared__ int ssrc[16], sdst[16], setp[16];
  __shared__ float satt[H_][16];
  const int tile = blockIdx.x, tid = threadIdx.x;
  if (tid < 16) {
    int e = tile * 16 + tid;
    ssrc[tid] = src[e]; sdst[tid] = dst[e]; setp[tid] = etype[e];
  }
  __syncthreads();
  const int h = tid >> 5, lane = tid & 31;
  const int m = lane & 15, half = lane >> 4;
  if (lane < 16) {
    int e = tile * 16 + lane;
    satt[h][lane] = exl[(size_t)e * H_ + h] / sm[(size_t)sdst[lane] * H_ + h];
  }
  __syncthreads();
  const int et = setp[m];
  v16bf a = load_a_bf(Vbf + (size_t)ssrc[m] * D_ + h * 32, half);
  const v16bf za = zero_frag();
  v8f c0 = {}, c1 = {};
  for (int r = 0; r < R_; ++r) {
    v16bf am = (et == r) ? a : za;
    const __bf16* b0p = relMT + (((size_t)(r * H_ + h) * 32) + m) * 32;
    v16bf b0 = load_b_bf(b0p, half);
    v16bf b1 = load_b_bf(b0p + 16 * 32, half);
    c0 = wmma_bf16(am, b0, c0);
    c1 = wmma_bf16(am, b1, c1);
  }
#pragma unroll
  for (int jj = 0; jj < 8; ++jj) {
    int mr = half * 8 + jj;
    int dd = sdst[mr];
    float at = satt[h][mr];
    atomicAdd(&aggr[(size_t)dd * D_ + h * 32 + m],      c0[jj] * at);
    atomicAdd(&aggr[(size_t)dd * D_ + h * 32 + 16 + m], c1[jj] * at);
  }
}

// ---- kernel 7: ELU + typed output linear (Wa) + residual -------------------
__global__ void __launch_bounds__(256)
update_kernel(const float* __restrict__ aggr, const float* __restrict__ x,
              const int* __restrict__ ntype, const __bf16* __restrict__ WaT,
              const float* __restrict__ ba, __bf16* __restrict__ Hbf) {
  __shared__ int sty[16];
  const int tile = blockIdx.x, tid = threadIdx.x;
  if (tid < 16) sty[tid] = ntype[tile * 16 + tid];
  __syncthreads();
  const int wave = tid >> 5, lane = tid & 31;
  const int m = lane & 15, half = lane >> 4;
  const int row = tile * 16 + m;
  const int tym = sty[m];
  v16bf afrag[8];
#pragma unroll
  for (int kc = 0; kc < 8; ++kc)
    afrag[kc] = load_a_elu(aggr + (size_t)row * D_ + kc * 32, half);
  const v16bf za = zero_frag();
#pragma unroll 1
  for (int j = 0; j < 2; ++j) {                  // 16 n-tiles / 8 waves
    const int nt = wave * 2 + j;
    const int col = nt * 16 + m;
    v8f c = {};
    for (int t = 0; t < T_; ++t) {
#pragma unroll
      for (int kc = 0; kc < 8; ++kc) {
        v16bf a2 = (tym == t) ? afrag[kc] : za;
        v16bf b = load_b_bf(WaT + (size_t)t * 65536 + (size_t)col * 256 + kc * 32, half);
        c = wmma_bf16(a2, b, c);
      }
    }
#pragma unroll
    for (int jj = 0; jj < 8; ++jj) {
      int mr = half * 8 + jj;
      int grow = tile * 16 + mr;
      float val = c[jj] + ba[(size_t)sty[mr] * D_ + col] +
                  x[(size_t)grow * D_ + col];
      Hbf[(size_t)grow * D_ + col] = (__bf16)val;
    }
  }
}

// ---- kernel 8: classifier GEMM + log_softmax (one wave per 16-row tile) ----
__global__ void __launch_bounds__(32)
classify_kernel(const __bf16* __restrict__ Hbf,
                const __bf16* __restrict__ WoutT,
                const float* __restrict__ bout, float* __restrict__ out) {
  __shared__ float sc[16][16];
  const int tile = blockIdx.x, lane = threadIdx.x;
  const int m = lane & 15, half = lane >> 4;
  const int row = tile * 16 + m;
  v8f c = {};
#pragma unroll
  for (int kc = 0; kc < 8; ++kc) {
    v16bf a = load_a_bf(Hbf + (size_t)row * D_ + kc * 32, half);
    v16bf b = load_b_bf(WoutT + (size_t)m * 256 + kc * 32, half);
    c = wmma_bf16(a, b, c);
  }
#pragma unroll
  for (int jj = 0; jj < 8; ++jj) {
    int mr = half * 8 + jj;
    sc[mr][m] = c[jj] + bout[m];
  }
  __syncthreads();
  if (lane < 16) {
    float vmax = -3.0e38f;
#pragma unroll
    for (int n = 0; n < C_; ++n) vmax = fmaxf(vmax, sc[lane][n]);
    float s = 0.f;
#pragma unroll
    for (int n = 0; n < C_; ++n) s += __expf(sc[lane][n] - vmax);
    float lse = vmax + __logf(s);
    int grow = tile * 16 + lane;
#pragma unroll
    for (int n = 0; n < C_; ++n) out[(size_t)grow * C_ + n] = sc[lane][n] - lse;
  }
}

// ---------------------------------------------------------------------------
extern "C" void kernel_launch(void* const* d_in, const int* in_sizes, int n_in,
                              void* d_out, int out_size, void* d_ws, size_t ws_size,
                              hipStream_t stream) {
  (void)in_sizes; (void)n_in; (void)out_size; (void)ws_size;
  const float* x     = (const float*)d_in[0];
  const int*   ei    = (const int*)d_in[1];
  const int*   ntype = (const int*)d_in[2];
  const int*   etype = (const int*)d_in[3];
  const float* Wk = (const float*)d_in[4];  const float* bk = (const float*)d_in[5];
  const float* Wq = (const float*)d_in[6];  const float* bq = (const float*)d_in[7];
  const float* Wv = (const float*)d_in[8];  const float* bv = (const float*)d_in[9];
  const float* Wa = (const float*)d_in[10]; const float* ba = (const float*)d_in[11];
  const float* rel_prior = (const float*)d_in[12];
  const float* rel_att   = (const float*)d_in[13];
  const float* rel_msg   = (const float*)d_in[14];
  const float* W_out     = (const float*)d_in[15];
  const float* b_out     = (const float*)d_in[16];
  const int* src = ei;
  const int* dst = ei + E_;

  char* ws = (char*)d_ws;
  __bf16* WT    = (__bf16*)(ws + OFF_WT);
  __bf16* relAT = (__bf16*)(ws + OFF_RELA);
  __bf16* relMT = (__bf16*)(ws + OFF_RELM);
  __bf16* WoutT = (__bf16*)(ws + OFF_WOUT);
  __bf16* Kbf   = (__bf16*)(ws + OFF_KBF);
  __bf16* Vbf   = (__bf16*)(ws + OFF_VBF);
  float* Qf   = (float*)(ws + OFF_QF);
  float* exl  = (float*)(ws + OFF_EXL);
  float* mx   = (float*)(ws + OFF_MX);
  float* sm   = (float*)(ws + OFF_SM);
  float* aggr = (float*)(ws + OFF_AGGR);
  __bf16* Hbf = (__bf16*)(ws + OFF_HBF);
  float* out = (float*)d_out;

  pack_kernel<<<3408, 256, 0, stream>>>(Wk, Wq, Wv, Wa, rel_att, rel_msg, W_out,
                                        WT, relAT, relMT, WoutT);
  init_kernel<<<(N_ * D_ + 255) / 256, 256, 0, stream>>>(mx, sm, aggr);
  qkv_kernel<<<N_ / 16, 256, 0, stream>>>(x, ntype, WT, bk, bq, bv, Kbf, Vbf, Qf);
  edge_logits_kernel<<<E_ / 16, 256, 0, stream>>>(src, dst, etype, Kbf, Qf, relAT,
                                                  rel_prior, exl);
  segmax_kernel<<<(E_ * H_ + 255) / 256, 256, 0, stream>>>(exl, dst, mx);
  segexp_kernel<<<(E_ * H_ + 255) / 256, 256, 0, stream>>>(exl, dst, mx, sm);
  edge_msg_kernel<<<E_ / 16, 256, 0, stream>>>(src, dst, etype, Vbf, relMT, exl, sm,
                                               aggr);
  update_kernel<<<N_ / 16, 256, 0, stream>>>(aggr, x, ntype,
                                             WT + (size_t)9 * 65536, ba, Hbf);
  classify_kernel<<<N_ / 16, 32, 0, stream>>>(Hbf, WoutT, b_out, out);
}
